// MoE_62483184222769
// MI455X (gfx1250) — matrix operand order repaired and checked by reading, not compiled, not run on hardware
//
#include <hip/hip_runtime.h>
#include <hip/hip_bf16.h>

// ---------------- MoE (SwiGLU, top-2 of 8 routed + 1 shared) for gfx1250 ----------------
// Gather-based dispatch (1 shared + 2 routed expert-passes of math instead of 9) with
// bf16 WMMA GEMMs (v_wmma_f32_16x16x32_bf16), native fp32->bf16 conversion,
// double-buffered LDS, async global->LDS copies for bf16 A tiles (ASYNCcnt path),
// register-staged B loads, and weight-tile prefetch into GL2.

#define T_TOK 8192
#define DIM   1024
#define FDIM  4096
#define NEXP  8

#define BM   128
#define BK   64     // K per LDS tile (2 WMMA k-steps)
#define B1N  64     // GEMM1 output (H) columns per block
#define B2N  128    // GEMM2 output columns per block
#define SAW  72     // LDS row stride in halves: 144B rows -> 16B aligned (b128), 16 distinct banks

#if defined(__has_builtin)
#if __has_builtin(__builtin_amdgcn_global_load_async_to_lds_b128) && \
    __has_builtin(__builtin_amdgcn_s_wait_asynccnt)
#define USE_ASYNC 1
#endif
#if __has_builtin(__builtin_amdgcn_cvt_pk_bf16_f32)
#define HAVE_CVTPK 1
#endif
#endif
#ifndef USE_ASYNC
#define USE_ASYNC 0
#endif
#ifndef HAVE_CVTPK
#define HAVE_CVTPK 0
#endif

typedef __attribute__((ext_vector_type(16))) __bf16       v16bf;
typedef __attribute__((ext_vector_type(2)))  __bf16       v2bf;
typedef __attribute__((ext_vector_type(8)))  float        v8f;
typedef __attribute__((ext_vector_type(8)))  unsigned int v8u;
typedef int v4i __attribute__((vector_size(16)));

// pack two floats into two bf16 (lo -> [15:0], hi -> [31:16])
__device__ __forceinline__ unsigned int pack2bf(float lo, float hi) {
#if HAVE_CVTPK
    v2bf r = __builtin_amdgcn_cvt_pk_bf16_f32(lo, hi);
    return __builtin_bit_cast(unsigned int, r);
#else
    v2bf r = { (__bf16)lo, (__bf16)hi };      // native fptrunc f32->bf16 (RNE)
    return __builtin_bit_cast(unsigned int, r);
#endif
}

__device__ __forceinline__ unsigned short f2bf(float f) {
    __bf16 b = (__bf16)f;
    return __builtin_bit_cast(unsigned short, b);
}

#if USE_ASYNC
__device__ __forceinline__ void async_cp_b128(const unsigned short* g, unsigned short* l) {
    __builtin_amdgcn_global_load_async_to_lds_b128((v4i*)g, (v4i*)l, 0, 0);
}
__device__ __forceinline__ void async_wait0() { __builtin_amdgcn_s_wait_asynccnt(0); }
#endif

// ---------------------------------------------------------------- small kernels
__global__ void zero_counts_k(int* __restrict__ counts) {
    if (threadIdx.x < NEXP) counts[threadIdx.x] = 0;
}

__global__ __launch_bounds__(256) void cvt_x_k(const float* __restrict__ x,
                                               unsigned short* __restrict__ xbf) {
    size_t i = ((size_t)blockIdx.x * 256 + threadIdx.x) * 4;   // exact cover: T*D % 1024 == 0
    float4 f = *(const float4*)(x + i);
    uint2 o;
    o.x = pack2bf(f.x, f.y);
    o.y = pack2bf(f.z, f.w);
    *(uint2*)(xbf + i) = o;
}

// one wave per token: logits -> softmax -> top2 -> combine weights + expert lists
__global__ __launch_bounds__(256) void gate_topk_k(const float* __restrict__ x,
                                                   const float* __restrict__ gw,
                                                   const float* __restrict__ gb,
                                                   float* __restrict__ combine,
                                                   int*   __restrict__ counts,
                                                   int*   __restrict__ idxl) {
    int lane = threadIdx.x & 31;
    int t    = blockIdx.x * 8 + (threadIdx.x >> 5);
    float acc[NEXP];
#pragma unroll
    for (int e = 0; e < NEXP; ++e) acc[e] = 0.f;
    for (int k = lane; k < DIM; k += 32) {
        float xv = x[(size_t)t * DIM + k];
        const float* g = gw + (size_t)k * NEXP;
#pragma unroll
        for (int e = 0; e < NEXP; ++e) acc[e] += xv * g[e];
    }
#pragma unroll
    for (int e = 0; e < NEXP; ++e)
        for (int off = 16; off; off >>= 1) acc[e] += __shfl_xor(acc[e], off, 32);
    if (lane == 0) {
        float lg[NEXP];
#pragma unroll
        for (int e = 0; e < NEXP; ++e) lg[e] = acc[e] + gb[e];
        int i1 = 0; float mx = lg[0];
#pragma unroll
        for (int e = 1; e < NEXP; ++e) if (lg[e] > mx) { mx = lg[e]; i1 = e; }
        float p[NEXP], sum = 0.f;
#pragma unroll
        for (int e = 0; e < NEXP; ++e) { p[e] = __expf(lg[e] - mx); sum += p[e]; }
        float inv = 1.0f / sum;
#pragma unroll
        for (int e = 0; e < NEXP; ++e) p[e] *= inv;
        int i2 = 0; float m2 = -1.f;
#pragma unroll
        for (int e = 0; e < NEXP; ++e) if (e != i1 && p[e] > m2) { m2 = p[e]; i2 = e; }
#pragma unroll
        for (int e = 0; e < NEXP; ++e)
            combine[(size_t)t * NEXP + e] = (e == i1) ? p[i1] : (e == i2) ? p[i2] : 0.0f;
        int p1 = atomicAdd(&counts[i1], 1); idxl[(size_t)i1 * T_TOK + p1] = t;
        int p2 = atomicAdd(&counts[i2], 1); idxl[(size_t)i2 * T_TOK + p2] = t;
    }
}

// ---------------------------------------------------------------- GEMM1: H = silu(x@w1a+b1a) * (x@w1b+b1b)
__global__ __launch_bounds__(256)
void gemm1_swiglu_k(const unsigned short* __restrict__ xbf,
                    const float* __restrict__ w1, const float* __restrict__ b1,
                    unsigned short* __restrict__ H,
                    const int* __restrict__ idxlist, const int* __restrict__ countp) {
    __shared__ unsigned short sA[2][BM * SAW];
    __shared__ unsigned short sB[2][2][B1N * SAW];

    int tokens = countp ? *countp : T_TOK;
    int m0 = blockIdx.x * BM;
    if (m0 >= tokens) return;
    int n0 = blockIdx.y * B1N;

    int tid = threadIdx.x;
    int lane = tid & 31, wave = tid >> 5;
    int wm = wave & 1, wn = wave >> 1;         // 2 (M) x 4 (N) wave grid, wave tile 64x16 (x2 sides)
    int lm = lane & 15, lh = lane >> 4;

    v8f accA[4], accB[4];
    v8f z = {0.f, 0.f, 0.f, 0.f, 0.f, 0.f, 0.f, 0.f};
#pragma unroll
    for (int i = 0; i < 4; ++i) { accA[i] = z; accB[i] = z; }

    float4 br[8];                               // B stage: 4 cells x (2 K rows of 4 floats)
    int arow = tid >> 1, ah = tid & 1;
    // A source row: out-of-range rows clamp to a valid token; their C rows are discarded later.
    int gr = m0 + arow;
    int cg = (gr < tokens) ? gr : (tokens - 1);
    int atok = idxlist ? idxlist[cg] : cg;
    const unsigned short* asrc = xbf + (size_t)atok * DIM + ah * 32;

#if USE_ASYNC
    auto load_A = [&](int kt, int buf) {
        const unsigned short* s = asrc + kt * BK;
        unsigned short* d = &sA[buf][arow * SAW + ah * 32];
#pragma unroll
        for (int i = 0; i < 4; ++i) async_cp_b128(s + i * 8, d + i * 8);
    };
#else
    uint4 ar[4];
    auto load_A = [&](int kt, int buf) {
        const uint4* p = (const uint4*)(asrc + kt * BK);
#pragma unroll
        for (int i = 0; i < 4; ++i) ar[i] = p[i];
        (void)buf;
    };
    auto store_A = [&](int buf) {
        unsigned int* dst = (unsigned int*)&sA[buf][arow * SAW + ah * 32];
#pragma unroll
        for (int i = 0; i < 4; ++i) {
            dst[4 * i + 0] = ar[i].x; dst[4 * i + 1] = ar[i].y;
            dst[4 * i + 2] = ar[i].z; dst[4 * i + 3] = ar[i].w;
        }
    };
#endif
    // B cells: (side, K-pair, 4 N). 2 sides x 32 k2 x 16 n4 = 1024 cells, 4 per thread.
    auto load_B = [&](int kt) {
        int k0 = kt * BK;
#pragma unroll
        for (int i = 0; i < 4; ++i) {
            int c = tid + i * 256;
            int s = c >> 9, cc = c & 511;
            int k = (cc >> 4) * 2, n = (cc & 15) * 4;
            const float* p = w1 + (size_t)(k0 + k) * (2 * FDIM) + (size_t)s * FDIM + n0 + n;
            br[2 * i]     = *(const float4*)p;
            br[2 * i + 1] = *(const float4*)(p + 2 * FDIM);
        }
    };
    auto store_B = [&](int buf) {
#pragma unroll
        for (int i = 0; i < 4; ++i) {
            int c = tid + i * 256;
            int s = c >> 9, cc = c & 511;
            int k = (cc >> 4) * 2, n = (cc & 15) * 4;
            unsigned short* base = &sB[buf][s][n * SAW + k];
            float4 f0 = br[2 * i], f1 = br[2 * i + 1];
            *(unsigned int*)(base + 0 * SAW) = pack2bf(f0.x, f1.x);
            *(unsigned int*)(base + 1 * SAW) = pack2bf(f0.y, f1.y);
            *(unsigned int*)(base + 2 * SAW) = pack2bf(f0.z, f1.z);
            *(unsigned int*)(base + 3 * SAW) = pack2bf(f0.w, f1.w);
        }
    };
    auto compute = [&](int buf) {
#pragma unroll
        for (int ks = 0; ks < 2; ++ks) {
#pragma unroll
            for (int mf = 0; mf < 4; ++mf) {
                int row = wm * 64 + mf * 16 + lm;
                v8u au;
#pragma unroll
                for (int v = 0; v < 8; ++v) {
                    int kb = ks * 32 + (v >> 2) * 16 + lh * 8 + (v & 3) * 2;
                    au[v] = *(const unsigned int*)&sA[buf][row * SAW + kb];
                }
                v16bf av = __builtin_bit_cast(v16bf, au);
#pragma unroll
                for (int s = 0; s < 2; ++s) {
                    int n = wn * 16 + lm;
                    v8u bu;
#pragma unroll
                    for (int v = 0; v < 8; ++v) {
                        int kb = ks * 32 + (v >> 2) * 16 + lh * 8 + (v & 3) * 2;
                        bu[v] = *(const unsigned int*)&sB[buf][s][n * SAW + kb];
                    }
                    v16bf bv = __builtin_bit_cast(v16bf, bu);
                    v8f& acc = s ? accB[mf] : accA[mf];
                    acc = __builtin_amdgcn_wmma_f32_16x16x32_bf16(false, av, false, bv,
                                                                  (short)0, acc, false, false);
                }
            }
        }
    };

    const int NK = DIM / BK;                        // 16
    load_A(0, 0);
    load_B(0);
#if !USE_ASYNC
    store_A(0);
#endif
    store_B(0);
#if USE_ASYNC
    async_wait0();
#endif
    __syncthreads();
    for (int kt = 0; kt < NK; ++kt) {
        if (kt + 1 < NK) { load_A(kt + 1, (kt + 1) & 1); load_B(kt + 1); }
        if (kt + 2 < NK) {                          // prefetch weight tile 2 steps ahead into GL2
            int k0 = (kt + 2) * BK;
            __builtin_prefetch(w1 + (size_t)(k0 + (tid >> 4)) * (2 * FDIM) + n0 + (tid & 15) * 4, 0, 0);
            __builtin_prefetch(w1 + (size_t)(k0 + (tid >> 4)) * (2 * FDIM) + FDIM + n0 + (tid & 15) * 4, 0, 0);
        }
        compute(kt & 1);
        if (kt + 1 < NK) {
            __syncthreads();
#if !USE_ASYNC
            store_A((kt + 1) & 1);
#endif
            store_B((kt + 1) & 1);
#if USE_ASYNC
            async_wait0();
#endif
            __syncthreads();
        }
    }
    // epilogue: bias + SwiGLU -> bf16 H (compacted rows)
    int nc = n0 + wn * 16 + lm;
    float ba = b1[nc], bb = b1[FDIM + nc];
#pragma unroll
    for (int mf = 0; mf < 4; ++mf) {
#pragma unroll
        for (int r = 0; r < 8; ++r) {
            int row = m0 + wm * 64 + mf * 16 + lh * 8 + r;
            if (row < tokens) {
                float a = accA[mf][r] + ba;
                float b = accB[mf][r] + bb;
                float h = (a / (1.0f + __expf(-a))) * b;
                H[(size_t)row * FDIM + nc] = f2bf(h);
            }
        }
    }
}

// ---------------------------------------------------------------- GEMM2: out[tok] (+)= (H@w2 + b2) * scale
__global__ __launch_bounds__(256)
void gemm2_out_k(const unsigned short* __restrict__ H,
                 const float* __restrict__ w2, const float* __restrict__ b2,
                 float* __restrict__ out,
                 const int* __restrict__ idxlist, const int* __restrict__ countp,
                 const float* __restrict__ combine, int expert, int overwrite) {
    __shared__ unsigned short sA[2][BM * SAW];
    __shared__ unsigned short sB[2][B2N * SAW];

    int tokens = countp ? *countp : T_TOK;
    int m0 = blockIdx.x * BM;
    if (m0 >= tokens) return;
    int n0 = blockIdx.y * B2N;

    int tid = threadIdx.x;
    int lane = tid & 31, wave = tid >> 5;
    int wm = wave & 1, wn = wave >> 1;         // wave tile 64(M) x 32(N)
    int lm = lane & 15, lh = lane >> 4;

    v8f acc[4][2];
    v8f z = {0.f, 0.f, 0.f, 0.f, 0.f, 0.f, 0.f, 0.f};
#pragma unroll
    for (int i = 0; i < 4; ++i) { acc[i][0] = z; acc[i][1] = z; }

    float4 br[8];
    int arow = tid >> 1, ah = tid & 1;
    int gr = m0 + arow;
    int cg = (gr < tokens) ? gr : (tokens - 1);     // H rows are compacted; clamp tail
    const unsigned short* asrc = H + (size_t)cg * FDIM + ah * 32;

#if USE_ASYNC
    auto load_A = [&](int kt, int buf) {
        const unsigned short* s = asrc + kt * BK;
        unsigned short* d = &sA[buf][arow * SAW + ah * 32];
#pragma unroll
        for (int i = 0; i < 4; ++i) async_cp_b128(s + i * 8, d + i * 8);
    };
#else
    uint4 ar[4];
    auto load_A = [&](int kt, int buf) {
        const uint4* p = (const uint4*)(asrc + kt * BK);
#pragma unroll
        for (int i = 0; i < 4; ++i) ar[i] = p[i];
        (void)buf;
    };
    auto store_A = [&](int buf) {
        unsigned int* dst = (unsigned int*)&sA[buf][arow * SAW + ah * 32];
#pragma unroll
        for (int i = 0; i < 4; ++i) {
            dst[4 * i + 0] = ar[i].x; dst[4 * i + 1] = ar[i].y;
            dst[4 * i + 2] = ar[i].z; dst[4 * i + 3] = ar[i].w;
        }
    };
#endif
    // B cells: (K-pair, 4 N). 32 k2 x 32 n4 = 1024 cells, 4 per thread.
    auto load_B = [&](int kt) {
        int k0 = kt * BK;
#pragma unroll
        for (int i = 0; i < 4; ++i) {
            int c = tid + i * 256;
            int k = (c >> 5) * 2, n = (c & 31) * 4;
            const float* p = w2 + (size_t)(k0 + k) * DIM + n0 + n;
            br[2 * i]     = *(const float4*)p;
            br[2 * i + 1] = *(const float4*)(p + DIM);
        }
    };
    auto store_B = [&](int buf) {
#pragma unroll
        for (int i = 0; i < 4; ++i) {
            int c = tid + i * 256;
            int k = (c >> 5) * 2, n = (c & 31) * 4;
            unsigned short* base = &sB[buf][n * SAW + k];
            float4 f0 = br[2 * i], f1 = br[2 * i + 1];
            *(unsigned int*)(base + 0 * SAW) = pack2bf(f0.x, f1.x);
            *(unsigned int*)(base + 1 * SAW) = pack2bf(f0.y, f1.y);
            *(unsigned int*)(base + 2 * SAW) = pack2bf(f0.z, f1.z);
            *(unsigned int*)(base + 3 * SAW) = pack2bf(f0.w, f1.w);
        }
    };
    auto compute = [&](int buf) {
#pragma unroll
        for (int ks = 0; ks < 2; ++ks) {
#pragma unroll
            for (int mf = 0; mf < 4; ++mf) {
                int row = wm * 64 + mf * 16 + lm;
                v8u au;
#pragma unroll
                for (int v = 0; v < 8; ++v) {
                    int kb = ks * 32 + (v >> 2) * 16 + lh * 8 + (v & 3) * 2;
                    au[v] = *(const unsigned int*)&sA[buf][row * SAW + kb];
                }
                v16bf av = __builtin_bit_cast(v16bf, au);
#pragma unroll
                for (int nf = 0; nf < 2; ++nf) {
                    int n = wn * 32 + nf * 16 + lm;
                    v8u bu;
#pragma unroll
                    for (int v = 0; v < 8; ++v) {
                        int kb = ks * 32 + (v >> 2) * 16 + lh * 8 + (v & 3) * 2;
                        bu[v] = *(const unsigned int*)&sB[buf][n * SAW + kb];
                    }
                    v16bf bv = __builtin_bit_cast(v16bf, bu);
                    acc[mf][nf] = __builtin_amdgcn_wmma_f32_16x16x32_bf16(false, av, false, bv,
                                                                          (short)0, acc[mf][nf],
                                                                          false, false);
                }
            }
        }
    };

    const int NK = FDIM / BK;                       // 64
    load_A(0, 0);
    load_B(0);
#if !USE_ASYNC
    store_A(0);
#endif
    store_B(0);
#if USE_ASYNC
    async_wait0();
#endif
    __syncthreads();
    for (int kt = 0; kt < NK; ++kt) {
        if (kt + 1 < NK) { load_A(kt + 1, (kt + 1) & 1); load_B(kt + 1); }
        if (kt + 2 < NK) {                          // prefetch weight tile 2 steps ahead into GL2
            int k0 = (kt + 2) * BK;
            __builtin_prefetch(w2 + (size_t)(k0 + (tid >> 3)) * DIM + n0 + (tid & 7) * 16, 0, 0);
        }
        compute(kt & 1);
        if (kt + 1 < NK) {
            __syncthreads();
#if !USE_ASYNC
            store_A((kt + 1) & 1);
#endif
            store_B((kt + 1) & 1);
#if USE_ASYNC
            async_wait0();
#endif
            __syncthreads();
        }
    }
    // epilogue
#pragma unroll
    for (int nf = 0; nf < 2; ++nf) {
        int nc = n0 + wn * 32 + nf * 16 + lm;
        float bias = b2[nc];
#pragma unroll
        for (int mf = 0; mf < 4; ++mf) {
#pragma unroll
            for (int r = 0; r < 8; ++r) {
                int row = m0 + wm * 64 + mf * 16 + lh * 8 + r;
                if (row < tokens) {
                    int tok = idxlist ? idxlist[row] : row;
                    float scale = combine ? combine[(size_t)tok * NEXP + expert] : 1.0f;
                    float v = (acc[mf][nf][r] + bias) * scale;
                    float* o = out + (size_t)tok * DIM + nc;
                    if (overwrite) *o = v; else *o += v;
                }
            }
        }
    }
}

// ---------------------------------------------------------------- launch
extern "C" void kernel_launch(void* const* d_in, const int* in_sizes, int n_in,
                              void* d_out, int out_size, void* d_ws, size_t ws_size,
                              hipStream_t stream) {
    (void)in_sizes; (void)n_in; (void)out_size; (void)ws_size;
    const float* x         = (const float*)d_in[0];
    const float* gate_w    = (const float*)d_in[1];
    const float* gate_b    = (const float*)d_in[2];
    const float* shared_w1 = (const float*)d_in[3];
    const float* shared_b1 = (const float*)d_in[4];
    const float* shared_w2 = (const float*)d_in[5];
    const float* shared_b2 = (const float*)d_in[6];
    const float* routed_w1 = (const float*)d_in[7];
    const float* routed_b1 = (const float*)d_in[8];
    const float* routed_w2 = (const float*)d_in[9];
    const float* routed_b2 = (const float*)d_in[10];
    float* out = (float*)d_out;

    char* ws = (char*)d_ws;
    size_t off = 0;
    auto alloc = [&](size_t b) { size_t o = off; off += (b + 255) & ~(size_t)255; return o; };
    float*          combine = (float*)(ws + alloc((size_t)T_TOK * NEXP * 4));
    int*            counts  = (int*)  (ws + alloc((size_t)NEXP * 4));
    int*            idxl    = (int*)  (ws + alloc((size_t)NEXP * T_TOK * 4));
    unsigned short* xbf     = (unsigned short*)(ws + alloc((size_t)T_TOK * DIM * 2));
    unsigned short* Hbuf    = (unsigned short*)(ws + alloc((size_t)T_TOK * FDIM * 2));

    zero_counts_k<<<1, 32, 0, stream>>>(counts);
    cvt_x_k<<<(T_TOK * DIM) / (256 * 4), 256, 0, stream>>>(x, xbf);
    gate_topk_k<<<T_TOK / 8, 256, 0, stream>>>(x, gate_w, gate_b, combine, counts, idxl);

    dim3 g1(T_TOK / BM, FDIM / B1N);
    dim3 g2(T_TOK / BM, DIM / B2N);

    // shared expert: dense, overwrite (initializes out)
    gemm1_swiglu_k<<<g1, 256, 0, stream>>>(xbf, shared_w1, shared_b1, Hbuf, nullptr, nullptr);
    gemm2_out_k<<<g2, 256, 0, stream>>>(Hbuf, shared_w2, shared_b2, out,
                                        nullptr, nullptr, nullptr, 0, 1);

    // routed experts: gathered token lists, weighted accumulate
    for (int e = 0; e < NEXP; ++e) {
        const float* w1 = routed_w1 + (size_t)e * DIM * 2 * FDIM;
        const float* b1 = routed_b1 + (size_t)e * 2 * FDIM;
        const float* w2 = routed_w2 + (size_t)e * FDIM * DIM;
        const float* b2 = routed_b2 + (size_t)e * DIM;
        gemm1_swiglu_k<<<g1, 256, 0, stream>>>(xbf, w1, b1, Hbuf, idxl + (size_t)e * T_TOK, counts + e);
        gemm2_out_k<<<g2, 256, 0, stream>>>(Hbuf, w2, b2, out,
                                            idxl + (size_t)e * T_TOK, counts + e, combine, e, 0);
    }
}
